// AMT_76209899700806
// MI455X (gfx1250) — matrix operand order repaired and checked by reading, not compiled, not run on hardware
//
#include <hip/hip_runtime.h>

// AMT BidirCorrBlock + _corr_scale_lookup for MI455X (gfx1250, wave32).
//
// Strategy: corr volume ops are linear in fmap1, so avgpool over the (u,v)
// axes of corr equals correlation against avgpooled features. We therefore:
//   1) convert/transpose features to [pixel][128] f16 (plus pooled levels 1-3),
//   2) run WMMA f16->f32 GEMMs: level-0 all-pairs volume (shared by both
//      directions) + 6 small pooled-level GEMMs (3 per direction),
//   3) one gather kernel does the 49-tap bilinear window lookups + flow concat.
// GEMM panel staging uses the Tensor Data Mover (tensor_load_to_lds) with
// hardware row padding + OOB zero-fill; falls back to manual staging if the
// TDM builtin is unavailable.

#define NPIX 6480    // 60*108
#define KD   128
#define HH   60
#define WW   108

typedef __attribute__((ext_vector_type(16))) _Float16     v16h;
typedef __attribute__((ext_vector_type(8)))  float        v8f;
typedef __attribute__((ext_vector_type(4)))  unsigned int v4u;
typedef __attribute__((ext_vector_type(4)))  unsigned int u32x4;
typedef __attribute__((ext_vector_type(8)))  int          i32x8;
typedef __attribute__((ext_vector_type(4)))  int          i32x4;

union Frag16 { v16h h; v4u q[2]; };

#if __has_builtin(__builtin_amdgcn_tensor_load_to_lds)
#define USE_TDM 1
#else
#define USE_TDM 0
#endif

// ---------------------------------------------------------------------------
// Feature transpose (+avg-pool for levels 1..3): [128][60][108] f32 ->
// [padN][128] f16, pixel-major / K-contiguous, zero-padded to padN pixels.
// ---------------------------------------------------------------------------
__global__ void amt_pool_feat(const float* __restrict__ in,  // [128][60][108]
                              _Float16* __restrict__ out,    // [padN][128]
                              int Hl, int Wl, int lvl) {
  const int pix = blockIdx.x;      // 0..padN-1
  const int k   = threadIdx.x;     // 0..127
  float v = 0.f;
  const int n = Hl * Wl;
  if (pix < n) {
    const int i = pix / Wl, j = pix - i * Wl;
    const int s = 1 << lvl;
    const float* base = in + (size_t)k * NPIX + (i * s) * WW + j * s;
    float sum = 0.f;
    for (int a = 0; a < s; ++a)
      for (int b = 0; b < s; ++b)
        sum += base[a * WW + b];
    v = sum / (float)(s * s);
  }
  out[(size_t)pix * KD + k] = (_Float16)v;
}

// ---------------------------------------------------------------------------
// Correlation GEMM: C[M=6480][padN] = scale * A(Mx128) * B(padNx128)^T
// A,B are [pixel][128] f16. Block: 9 waves (288 thr), tile 144x64, K=128 in
// LDS (row stride 136 halves = 272B => conflict-free b128 fragment reads).
// Each wave: 16x64 strip, 4 K-steps x 4 N-tiles = 16 v_wmma_f32_16x16x32_f16.
// Panels staged by TDM: 2D tile, 32x8B per row, hardware pad 16B per 256B row
// (pad_interval code 5 = 64 DW, pad_amount code 3 = 4 DW), OOB rows -> zeros.
// ---------------------------------------------------------------------------
#define BM  144
#define BN  64
#define LDA 136   // halves per LDS row (128 + 8 pad = 272B)

#if USE_TDM
__device__ __forceinline__ void tdm_load_panel(const _Float16* gsrc,
                                               unsigned lds_byte_off,
                                               int tensor_rows,
                                               int tile_rows) {
  const unsigned long long ga = (unsigned long long)gsrc;
  u32x4 g0;
  g0[0] = 1u;                                   // count=1 (valid descriptor)
  g0[1] = lds_byte_off;                         // lds_addr [63:32]
  g0[2] = (unsigned)(ga & 0xffffffffu);         // global_addr lo
  g0[3] = (unsigned)((ga >> 32) & 0x01ffffffu)  // global_addr [56:32]
          | (2u << 30);                         // type = 2 ("image")
  i32x8 g1;
  g1[0] = (3 << 16)      // data_size = 3 (8 bytes)
        | (1 << 20)      // pad_enable
        | (5 << 22)      // pad_interval code 5 = 64 DWORDs (256B)
        | (3 << 25);     // pad_amount  code 3 = 4 DWORDs (16B)
  g1[1] = (32 & 0xffff) << 16;                        // tensor_dim0 lo16 (32 x 8B)
  g1[2] = 0 | ((tensor_rows & 0xffff) << 16);         // dim0 hi16 | tensor_dim1 lo16
  g1[3] = ((tensor_rows >> 16) & 0xffff) | (32 << 16);// dim1 hi16 | tile_dim0 = 32
  g1[4] = (tile_rows & 0xffff);                       // tile_dim1 | tile_dim2=0
  g1[5] = 32;                                         // tensor_dim0_stride = 32 (8B units)
  g1[6] = 0;
  g1[7] = 0;
  i32x4 gz4 = {0, 0, 0, 0};
  i32x8 gz8 = {0, 0, 0, 0, 0, 0, 0, 0};
  // amdgpu-toolchain (clang-23) 6-arg form.
  __builtin_amdgcn_tensor_load_to_lds(g0, g1, gz4, gz4, gz8, 0);
}
#endif

__global__ __launch_bounds__(288)
void amt_corr_wmma(const _Float16* __restrict__ A,
                   const _Float16* __restrict__ B,
                   float* __restrict__ C,
                   int padN, float scale) {
  __shared__ alignas(16) _Float16 lA[BM * LDA];
  __shared__ alignas(16) _Float16 lB[BN * LDA];
  const int tid     = threadIdx.x;
  const int rowBase = blockIdx.y * BM;   // M = 6480 = 45*144, always exact
  const int colBase = blockIdx.x * BN;

#if USE_TDM
  if (tid < 32) {  // wave 0 issues the DMAs (TDM ignores EXEC; branch-guarded)
    // Generic LDS pointers live in the LDS aperture: addr[31:0] is the LDS
    // byte offset (ISA 10.2 aperture mapping).
    const unsigned ldsA = (unsigned)(unsigned long long)(const void*)&lA[0];
    const unsigned ldsB = (unsigned)(unsigned long long)(const void*)&lB[0];
    tdm_load_panel(A + (size_t)rowBase * KD, ldsA, BM, BM);
    // tensor_dim1 = remaining rows: TDM zero-fills OOB rows => B-panel pad.
    tdm_load_panel(B + (size_t)colBase * KD, ldsB, padN - colBase, BN);
#if __has_builtin(__builtin_amdgcn_s_wait_tensorcnt)
    __builtin_amdgcn_s_wait_tensorcnt(0);
#else
    asm volatile("s_wait_tensorcnt 0x0" ::: "memory");
#endif
  }
#else
  // Stage A panel: 144 rows x 256B = 2304 16B-chunks / 288 threads = 8 each.
  {
    const v4u* ga = (const v4u*)(A + (size_t)rowBase * KD);
#pragma unroll
    for (int i = 0; i < 8; ++i) {
      const int idx = tid + i * 288;
      const int r = idx >> 4, c = idx & 15;
      *(v4u*)(&lA[r * LDA + c * 8]) = ga[r * 16 + c];
    }
  }
  // Stage B panel: 64 rows -> 1024 chunks, guarded + zero-filled past padN.
#pragma unroll
  for (int i = 0; i < 4; ++i) {
    const int idx = tid + i * 288;
    if (idx < 1024) {
      const int r = idx >> 4, c = idx & 15;
      const int q = colBase + r;
      v4u d = {};
      if (q < padN) d = *(const v4u*)(B + (size_t)q * KD + c * 8);
      *(v4u*)(&lB[r * LDA + c * 8]) = d;
    }
  }
#endif
  __syncthreads();

  const int wave = tid >> 5;
  const int lane = tid & 31;
  const int l16  = lane & 15;
  const int hi   = lane >> 4;

  v8f zero = {};
  v8f acc[4];
#pragma unroll
  for (int t = 0; t < 4; ++t) acc[t] = zero;

#pragma unroll
  for (int s = 0; s < 4; ++s) {
    const int kb = s * 32;
    // A fragment (ISA 16-bit 16x32 layout): halves[0..7]=K[kb+hi*8 ..],
    // halves[8..15]=K[kb+16+hi*8 ..], row = wave*16 + (lane&15).
    Frag16 fa;
    const _Float16* pa = &lA[(wave * 16 + l16) * LDA + kb + hi * 8];
    fa.q[0] = *(const v4u*)pa;
    fa.q[1] = *(const v4u*)(pa + 16);
#pragma unroll
    for (int t = 0; t < 4; ++t) {
      // B fragment: halves[0..7]=K[kb+hi*8 ..], halves[8..15]=K[kb+16+hi*8 ..],
      // col = t*16 + (lane&15).
      Frag16 fb;
      const _Float16* pb = &lB[(t * 16 + l16) * LDA + kb + hi * 8];
      fb.q[0] = *(const v4u*)pb;
      fb.q[1] = *(const v4u*)(pb + 16);
      acc[t] = __builtin_amdgcn_wmma_f32_16x16x32_f16(
          false, fa.h, false, fb.h, (short)0, acc[t], false, false);
    }
  }

  // C/D 16x16 f32 layout: VGPR j -> M = j + hi*8, N = lane&15.
  const int row0 = rowBase + wave * 16 + hi * 8;
#pragma unroll
  for (int t = 0; t < 4; ++t) {
    const int col = colBase + t * 16 + l16;
    if (col < padN) {
      float* cp = C + (size_t)row0 * padN + col;
#pragma unroll
      for (int j = 0; j < 8; ++j)
        cp[(size_t)j * padN] = acc[t][j] * scale;
    }
  }
}

// ---------------------------------------------------------------------------
// 49-tap bilinear window lookup over 4 levels, both directions, + flows.
// Output layout [396][6480]: ch = dir*196 + l*49 + t  (t: dy outer, dx inner),
// ch 392..395 = flow0(2), flow1(2). Dir-1 level-0 reads corr0 transposed.
// ---------------------------------------------------------------------------
__global__ void amt_lookup(const float* __restrict__ flow0,
                           const float* __restrict__ flow1,
                           const float* __restrict__ embt,
                           const float* __restrict__ corr0,
                           const float* __restrict__ c0l1,
                           const float* __restrict__ c0l2,
                           const float* __restrict__ c0l3,
                           const float* __restrict__ c1l1,
                           const float* __restrict__ c1l2,
                           const float* __restrict__ c1l3,
                           float* __restrict__ out) {
  const int gid = blockIdx.x * blockDim.x + threadIdx.x;
  const int total = 396 * NPIX;
  if (gid >= total) return;
  const int ch = gid / NPIX;
  const int p  = gid - ch * NPIX;

  if (ch >= 392) {
    const int f = ch - 392;
    out[gid] = (f < 2) ? flow0[f * NPIX + p] : flow1[(f - 2) * NPIX + p];
    return;
  }

  const int dir = ch / 196;
  const int rem = ch - dir * 196;
  const int l   = rem / 49;
  const int t   = rem - l * 49;
  const float dy = (float)(t / 7) - 3.f;
  const float dx = (float)(t % 7) - 3.f;

  const float e = embt[0];
  float fs, fx, fy;
  if (dir == 0) { fs = 1.f / e;          fx = flow1[p]; fy = flow1[NPIX + p]; }
  else          { fs = 1.f / (1.f - e);  fx = flow0[p]; fy = flow0[NPIX + p]; }

  const float inv = 1.f / (float)(1 << l);
  const float x = ((float)(p % WW) + fx * fs) * inv + dx;
  const float y = ((float)(p / WW) + fy * fs) * inv + dy;

  const int HLt[4] = {60, 30, 15, 7};
  const int WLt[4] = {108, 54, 27, 13};
  const int PNt[4] = {6480, 1632, 416, 96};   // padded row strides
  const int hl = HLt[l], wl = WLt[l], pn = PNt[l];

  const float x0f = floorf(x), y0f = floorf(y);
  const float wx1 = x - x0f, wx0 = 1.f - wx1;
  const float wy1 = y - y0f, wy0 = 1.f - wy1;
  const int ix = (int)x0f, iy = (int)y0f;

  const float* buf;
  if (dir == 0) buf = (l == 0) ? corr0 : (l == 1) ? c0l1 : (l == 2) ? c0l2 : c0l3;
  else          buf = (l == 0) ? corr0 : (l == 1) ? c1l1 : (l == 2) ? c1l2 : c1l3;
  const bool transposed = (dir == 1) && (l == 0);

  auto fetch = [&](int qx, int qy) -> float {
    if (qx < 0 || qx >= wl || qy < 0 || qy >= hl) return 0.f;
    const size_t idx = transposed ? ((size_t)(qy * WW + qx) * NPIX + p)
                                  : ((size_t)p * pn + (size_t)(qy * wl + qx));
    return buf[idx];
  };

  out[gid] = fetch(ix,     iy    ) * wx0 * wy0 +
             fetch(ix + 1, iy    ) * wx1 * wy0 +
             fetch(ix,     iy + 1) * wx0 * wy1 +
             fetch(ix + 1, iy + 1) * wx1 * wy1;
}

// ---------------------------------------------------------------------------
extern "C" void kernel_launch(void* const* d_in, const int* in_sizes, int n_in,
                              void* d_out, int out_size, void* d_ws, size_t ws_size,
                              hipStream_t stream) {
  const float* fmap0 = (const float*)d_in[0];
  const float* fmap1 = (const float*)d_in[1];
  const float* flow0 = (const float*)d_in[2];
  const float* flow1 = (const float*)d_in[3];
  const float* embt  = (const float*)d_in[4];
  float* out = (float*)d_out;

  size_t off = 0;
  auto take = [&](size_t bytes) -> void* {
    void* p = (char*)d_ws + off;
    off += (bytes + 255) & ~(size_t)255;
    return p;
  };

  _Float16* f0t = (_Float16*)take((size_t)NPIX * KD * 2);
  _Float16* f1t = (_Float16*)take((size_t)NPIX * KD * 2);
  const int PN[3] = {1632, 416, 96};    // pooled levels padded to mult. of 16
  const int HL[3] = {30, 15, 7};
  const int WL[3] = {54, 27, 13};
  _Float16 *f0p[3], *f1p[3];
  for (int l = 0; l < 3; ++l) f1p[l] = (_Float16*)take((size_t)PN[l] * KD * 2);
  for (int l = 0; l < 3; ++l) f0p[l] = (_Float16*)take((size_t)PN[l] * KD * 2);
  float* corr0 = (float*)take((size_t)NPIX * NPIX * 4);     // 168 MB
  float *c0[3], *c1[3];
  for (int l = 0; l < 3; ++l) c0[l] = (float*)take((size_t)NPIX * PN[l] * 4);
  for (int l = 0; l < 3; ++l) c1[l] = (float*)take((size_t)NPIX * PN[l] * 4);

  // Feature transpose + pooled pyramids (avgpool commutes with the dot).
  amt_pool_feat<<<NPIX, 128, 0, stream>>>(fmap0, f0t, HH, WW, 0);
  amt_pool_feat<<<NPIX, 128, 0, stream>>>(fmap1, f1t, HH, WW, 0);
  for (int l = 0; l < 3; ++l) {
    amt_pool_feat<<<PN[l], 128, 0, stream>>>(fmap1, f1p[l], HL[l], WL[l], l + 1);
    amt_pool_feat<<<PN[l], 128, 0, stream>>>(fmap0, f0p[l], HL[l], WL[l], l + 1);
  }

  const float scale = 0.08838834764831845f;   // 1/sqrt(128)
  auto gemm = [&](const _Float16* Af, const _Float16* Bf, float* Cf, int padN) {
    dim3 g((padN + BN - 1) / BN, NPIX / BM);
    amt_corr_wmma<<<g, 288, 0, stream>>>(Af, Bf, Cf, padN, scale);
  };
  gemm(f0t, f1t, corr0, NPIX);                       // level-0 all-pairs volume
  for (int l = 0; l < 3; ++l) gemm(f0t, f1p[l], c0[l], PN[l]);  // dir 0 pooled
  for (int l = 0; l < 3; ++l) gemm(f1t, f0p[l], c1[l], PN[l]);  // dir 1 pooled

  const int total = 396 * NPIX;
  amt_lookup<<<(total + 255) / 256, 256, 0, stream>>>(
      flow0, flow1, embt, corr0,
      c0[0], c0[1], c0[2], c1[0], c1[1], c1[2], out);
}
